// MultiheadAttention_13185549599116
// MI455X (gfx1250) — matrix operand order repaired and checked by reading, not compiled, not run on hardware
//
#include <hip/hip_runtime.h>

// ---------------- CDNA5 WMMA types ----------------
typedef __bf16 bf16_t;
typedef __attribute__((ext_vector_type(8)))  __bf16 v8bf;
typedef __attribute__((ext_vector_type(16))) __bf16 v16bf;
typedef __attribute__((ext_vector_type(8)))  float  v8f;

#define WMMA_BF16(a, b, c) \
  __builtin_amdgcn_wmma_f32_16x16x32_bf16(false, (a), false, (b), (short)0, (c), false, false)

// Problem constants
constexpr int Bc = 4, Tc = 2048, Dc = 1024, Hc = 16, HDc = 64;

__device__ __forceinline__ bf16_t f2bf(float f) { return (bf16_t)f; }

// A-fragment 16x32 (row-major src, leading dim ld elements), tile origin (row0, k0).
// Lane layout (ISA 7.12.2): row m = lane%16; K chunks [8*half, +8) and [16+8*half, +8).
__device__ __forceinline__ v16bf load_a_frag(const bf16_t* src, int ld, int row0, int k0) {
  int lane = threadIdx.x & 31;
  int m = lane & 15, half = lane >> 4;
  const bf16_t* p = src + (size_t)(row0 + m) * ld + k0 + half * 8;
  v8bf c0 = *(const v8bf*)p;
  v8bf c1 = *(const v8bf*)(p + 16);
  v16bf a;
#pragma unroll
  for (int i = 0; i < 8; ++i) { a[i] = c0[i]; a[i + 8] = c1[i]; }
  return a;
}

// B-fragment 32x16 (row-major src): lane holds column N = n0+lane%16; K chunks [8*half, +8) and [16+8*half, +8).
__device__ __forceinline__ v16bf load_b_frag(const bf16_t* src, int ld, int k0, int n0) {
  int lane = threadIdx.x & 31;
  int n = lane & 15, half = lane >> 4;
  const bf16_t* p = src + (size_t)(k0 + half * 8) * ld + n0 + n;
  v16bf b;
#pragma unroll
  for (int i = 0; i < 8; ++i) { b[i] = p[(size_t)i * ld]; b[i + 8] = p[(size_t)(i + 16) * ld]; }
  return b;
}

// ---------------- f32 -> bf16 convert (8 elems / thread) ----------------
__global__ void cvt_bf16_kernel(const float* __restrict__ in, bf16_t* __restrict__ out, int n) {
  int i = (blockIdx.x * blockDim.x + threadIdx.x) * 8;
  if (i < n) {
    float4 f0 = *(const float4*)(in + i);
    float4 f1 = *(const float4*)(in + i + 4);
    v8bf o;
    o[0] = f2bf(f0.x); o[1] = f2bf(f0.y); o[2] = f2bf(f0.z); o[3] = f2bf(f0.w);
    o[4] = f2bf(f1.x); o[5] = f2bf(f1.y); o[6] = f2bf(f1.z); o[7] = f2bf(f1.w);
    *(v8bf*)(out + i) = o;
  }
}

// ---------------- fused QKV projection (software pipelined) ----------------
// One wave: 16 rows x 64 cols (= one head) of q,k,v. While the 4 WMMAs of
// weight w run, the B-fragments of weight w+1 (and next A at k-step edges)
// are in flight. Prefetch indices wrap (& Dc-1) => branchless, in-bounds.
__global__ void __launch_bounds__(128) qkv_proj_kernel(
    const bf16_t* __restrict__ xb, const bf16_t* __restrict__ wq,
    const bf16_t* __restrict__ wk, const bf16_t* __restrict__ wv,
    bf16_t* __restrict__ Qb, bf16_t* __restrict__ Ktb, bf16_t* __restrict__ Vb) {
  int lane = threadIdx.x & 31;
  int wid = blockIdx.x * 4 + (threadIdx.x >> 5);   // 8192 waves
  int rt = wid >> 4;                               // 512 row tiles of 16
  int h  = wid & 15;                               // head = 64-wide col tile
  int row0 = rt * 16;
  int n0 = h * HDc;

  const bf16_t* const Ws[3] = {wq, wk, wv};
  v8f c[3][4] = {};

  v16bf a_cur = load_a_frag(xb, Dc, row0, 0);
  v16bf b_cur[4];
#pragma unroll
  for (int nf = 0; nf < 4; ++nf) b_cur[nf] = load_b_frag(wq, Dc, 0, n0 + nf * 16);

  for (int k0 = 0; k0 < Dc; k0 += 32) {
#pragma unroll
    for (int w = 0; w < 3; ++w) {
      const bf16_t* nsrc = Ws[(w + 1) % 3];
      int nk0 = (w == 2) ? ((k0 + 32) & (Dc - 1)) : k0;
      v16bf b_nxt[4];
#pragma unroll
      for (int nf = 0; nf < 4; ++nf) b_nxt[nf] = load_b_frag(nsrc, Dc, nk0, n0 + nf * 16);
      v16bf a_nxt = a_cur;
      if (w == 2) a_nxt = load_a_frag(xb, Dc, row0, nk0);
#pragma unroll
      for (int nf = 0; nf < 4; ++nf) c[w][nf] = WMMA_BF16(a_cur, b_cur[nf], c[w][nf]);
#pragma unroll
      for (int nf = 0; nf < 4; ++nf) b_cur[nf] = b_nxt[nf];
      a_cur = a_nxt;
    }
  }

  int b_idx = row0 >> 11;          // row0 / T
  int t0 = row0 & (Tc - 1);
  int m = lane & 15, half = lane >> 4;
  size_t base = (size_t)(b_idx * Hc + h) * Tc * HDc;   // same base for Q/Kt/V
#pragma unroll
  for (int nf = 0; nf < 4; ++nf) {
#pragma unroll
    for (int r = 0; r < 8; ++r) {
      int t = t0 + r + 8 * half;
      int hd = nf * 16 + m;
      Qb [base + (size_t)t * HDc + hd] = f2bf(c[0][nf][r]);
      Ktb[base + (size_t)hd * Tc + t]  = f2bf(c[1][nf][r]);
      Vb [base + (size_t)t * HDc + hd] = f2bf(c[2][nf][r]);
    }
  }
}

// ---------------- flash-style causal attention ----------------
// One wave = 16 queries of one (b,h). Streams 32-key tiles with K^T
// fragments double-buffered across iterations; V fragments issued right
// after the score WMMAs so softmax VALU work hides their latency.
__device__ __forceinline__ void load_k_frags(const bf16_t* Kt, int kc, v16bf kb[4]) {
  kb[0] = load_b_frag(Kt, Tc, 0,  kc);
  kb[1] = load_b_frag(Kt, Tc, 32, kc);
  kb[2] = load_b_frag(Kt, Tc, 0,  kc + 16);
  kb[3] = load_b_frag(Kt, Tc, 32, kc + 16);
}

__global__ void __launch_bounds__(128) attn_kernel(
    const bf16_t* __restrict__ Qb, const bf16_t* __restrict__ Ktb,
    const bf16_t* __restrict__ Vb, bf16_t* __restrict__ Zb) {
  __shared__ bf16_t pshare[4][16 * 32];   // 1KB per wave
  int lane = threadIdx.x & 31;
  int wslot = threadIdx.x >> 5;
  int wid = blockIdx.x * 4 + wslot;       // 8192 waves
  int qt = wid & 127;                     // T/16 query tiles
  int bh = wid >> 7;                      // 0..63
  int q0 = qt * 16;

  const bf16_t* Q  = Qb  + (size_t)bh * Tc * HDc;
  const bf16_t* Kt = Ktb + (size_t)bh * Tc * HDc;
  const bf16_t* V  = Vb  + (size_t)bh * Tc * HDc;
  bf16_t* P = &pshare[wslot][0];

  int m = lane & 15, half = lane >> 4;
  const float sm_scale = 0.125f;          // 1/sqrt(64)

  v16bf qa0 = load_a_frag(Q, HDc, q0, 0);
  v16bf qa1 = load_a_frag(Q, HDc, q0, 32);

  v8f acc[4] = {};
  float mi[8], li[8];
#pragma unroll
  for (int r = 0; r < 8; ++r) { mi[r] = -3.0e38f; li[r] = 0.0f; }

  v16bf kb[4];
  load_k_frags(Kt, 0, kb);

  for (int k0 = 0; k0 < q0 + 16; k0 += 32) {
    // ---- scores: two 16-key fragments, hd reduced in two 32-chunks ----
    v8f s0 = {}; v8f s1 = {};
    s0 = WMMA_BF16(qa0, kb[0], s0);
    s0 = WMMA_BF16(qa1, kb[1], s0);
    s1 = WMMA_BF16(qa0, kb[2], s1);
    s1 = WMMA_BF16(qa1, kb[3], s1);

    // ---- prefetch V (this tile) and K^T (next tile, wrapped) ----
    v16bf vb[4];
#pragma unroll
    for (int nf = 0; nf < 4; ++nf) vb[nf] = load_b_frag(V, HDc, k0, nf * 16);
    v16bf kn[4];
    load_k_frags(Kt, (k0 + 32) & (Tc - 1), kn);

    // ---- causal mask + online softmax (rows live in 16-lane halves) ----
    float scl[8];
#pragma unroll
    for (int r = 0; r < 8; ++r) {
      int row = q0 + r + 8 * half;
      float v0 = (k0 + m      <= row) ? s0[r] * sm_scale : -3.0e38f;
      float v1 = (k0 + 16 + m <= row) ? s1[r] * sm_scale : -3.0e38f;
      float tmax = fmaxf(v0, v1);
#pragma unroll
      for (int off = 1; off <= 8; off <<= 1)
        tmax = fmaxf(tmax, __shfl_xor(tmax, off, 32));
      float nm = fmaxf(mi[r], tmax);
      float sc = __expf(mi[r] - nm);
      float p0 = __expf(v0 - nm);
      float p1 = __expf(v1 - nm);
      float rs = p0 + p1;
#pragma unroll
      for (int off = 1; off <= 8; off <<= 1)
        rs += __shfl_xor(rs, off, 32);
      li[r] = li[r] * sc + rs;
      mi[r] = nm;
      scl[r] = sc;
      // P (16x32) to LDS in plain row-major for C->A relayout
      P[(r + 8 * half) * 32 + m]      = f2bf(p0);
      P[(r + 8 * half) * 32 + 16 + m] = f2bf(p1);
    }
#pragma unroll
    for (int nf = 0; nf < 4; ++nf)
#pragma unroll
      for (int r = 0; r < 8; ++r)
        acc[nf][r] *= scl[r];

    __asm__ volatile("s_wait_dscnt 0" ::: "memory");

    // ---- read P back as A-fragment (ld = 32) ----
    v16bf pa;
    {
      const bf16_t* pp = P + m * 32 + half * 8;
      v8bf c0 = *(const v8bf*)pp;
      v8bf c1 = *(const v8bf*)(pp + 16);
#pragma unroll
      for (int i = 0; i < 8; ++i) { pa[i] = c0[i]; pa[i + 8] = c1[i]; }
    }

    // ---- acc += P V (V row-major [T,64] is already B-friendly) ----
#pragma unroll
    for (int nf = 0; nf < 4; ++nf)
      acc[nf] = WMMA_BF16(pa, vb[nf], acc[nf]);

#pragma unroll
    for (int i = 0; i < 4; ++i) kb[i] = kn[i];
  }

  // ---- normalize and emit z (bf16, [B,T,D] with D = h*64+hd) ----
  int b_idx = bh >> 4, h = bh & 15;
#pragma unroll
  for (int r = 0; r < 8; ++r) {
    float inv = 1.0f / li[r];
    size_t row = (size_t)b_idx * Tc + q0 + r + 8 * half;
#pragma unroll
    for (int nf = 0; nf < 4; ++nf)
      Zb[row * Dc + h * HDc + nf * 16 + m] = f2bf(acc[nf][r] * inv);
  }
}

// ---------------- output projection: out = Z Wo + bo (f32 out) ----------------
__global__ void __launch_bounds__(128) out_proj_kernel(
    const bf16_t* __restrict__ Zb, const bf16_t* __restrict__ wo,
    const float* __restrict__ bo, float* __restrict__ out) {
  int lane = threadIdx.x & 31;
  int wid = blockIdx.x * 4 + (threadIdx.x >> 5);
  int rt = wid >> 4;
  int nt = wid & 15;
  int row0 = rt * 16, n0 = nt * 64;

  v8f c[4] = {};
  v16bf a_cur = load_a_frag(Zb, Dc, row0, 0);
  v16bf b_cur[4];
#pragma unroll
  for (int nf = 0; nf < 4; ++nf) b_cur[nf] = load_b_frag(wo, Dc, 0, n0 + nf * 16);

  for (int k0 = 0; k0 < Dc; k0 += 32) {
    int nk0 = (k0 + 32) & (Dc - 1);
    v16bf b_nxt[4];
#pragma unroll
    for (int nf = 0; nf < 4; ++nf) b_nxt[nf] = load_b_frag(wo, Dc, nk0, n0 + nf * 16);
    v16bf a_nxt = load_a_frag(Zb, Dc, row0, nk0);
#pragma unroll
    for (int nf = 0; nf < 4; ++nf) c[nf] = WMMA_BF16(a_cur, b_cur[nf], c[nf]);
#pragma unroll
    for (int nf = 0; nf < 4; ++nf) b_cur[nf] = b_nxt[nf];
    a_cur = a_nxt;
  }

  int m = lane & 15, half = lane >> 4;
#pragma unroll
  for (int nf = 0; nf < 4; ++nf) {
    float bias = bo[n0 + nf * 16 + m];
#pragma unroll
    for (int r = 0; r < 8; ++r)
      out[(size_t)(row0 + r + 8 * half) * Dc + n0 + nf * 16 + m] = c[nf][r] + bias;
  }
}

// ---------------- launcher ----------------
extern "C" void kernel_launch(void* const* d_in, const int* in_sizes, int n_in,
                              void* d_out, int out_size, void* d_ws, size_t ws_size,
                              hipStream_t stream) {
  (void)in_sizes; (void)n_in; (void)out_size; (void)ws_size;
  const float* x  = (const float*)d_in[0];
  const float* Wq = (const float*)d_in[1];
  const float* Wk = (const float*)d_in[2];
  const float* Wv = (const float*)d_in[3];
  const float* Wo = (const float*)d_in[4];
  const float* bo = (const float*)d_in[5];
  float* out = (float*)d_out;

  char* ws = (char*)d_ws;
  const size_t MB = (size_t)1 << 20;
  bf16_t* xb  = (bf16_t*)(ws + 0 * MB);   // 16 MB
  bf16_t* wqb = (bf16_t*)(ws + 16 * MB);  // 2 MB
  bf16_t* wkb = (bf16_t*)(ws + 18 * MB);  // 2 MB
  bf16_t* wvb = (bf16_t*)(ws + 20 * MB);  // 2 MB
  bf16_t* wob = (bf16_t*)(ws + 22 * MB);  // 2 MB
  bf16_t* Qb  = (bf16_t*)(ws + 24 * MB);  // 16 MB  [B,H,T,64]
  bf16_t* Ktb = (bf16_t*)(ws + 40 * MB);  // 16 MB  [B,H,64,T]
  bf16_t* Vb  = (bf16_t*)(ws + 56 * MB);  // 16 MB  [B,H,T,64]
  bf16_t* Zb  = (bf16_t*)(ws + 72 * MB);  // 16 MB  [B,T,D]

  const int nx = Bc * Tc * Dc;   // 8,388,608
  const int nw = Dc * Dc;        // 1,048,576
  cvt_bf16_kernel<<<(nx / 8 + 255) / 256, 256, 0, stream>>>(x,  xb,  nx);
  cvt_bf16_kernel<<<(nw / 8 + 255) / 256, 256, 0, stream>>>(Wq, wqb, nw);
  cvt_bf16_kernel<<<(nw / 8 + 255) / 256, 256, 0, stream>>>(Wk, wkb, nw);
  cvt_bf16_kernel<<<(nw / 8 + 255) / 256, 256, 0, stream>>>(Wv, wvb, nw);
  cvt_bf16_kernel<<<(nw / 8 + 255) / 256, 256, 0, stream>>>(Wo, wob, nw);

  // 8192 waves each: (B*T/16 row tiles) x (H col tiles) ; 4 waves / block
  qkv_proj_kernel<<<2048, 128, 0, stream>>>(xb, wqb, wkb, wvb, Qb, Ktb, Vb);
  // B*H*(T/16) = 8192 waves ; 4 waves / block
  attn_kernel<<<2048, 128, 0, stream>>>(Qb, Ktb, Vb, Zb);
  out_proj_kernel<<<2048, 128, 0, stream>>>(Zb, wob, bo, out);
}
